// EGNN_5145370820929
// MI455X (gfx1250) — matrix operand order repaired and checked by reading, not compiled, not run on hardware
//
#include <hip/hip_runtime.h>
#include <hip/hip_bf16.h>

typedef __bf16 bf16_t;
typedef __attribute__((ext_vector_type(16))) __bf16 v16bf;
typedef __attribute__((ext_vector_type(8)))  float  v8f;

#define N_NODES 512
#define DIM 128
#define MD 32
#define H1 514
#define H1P 544
#define KSTEPS 17

// workspace byte offsets (all 32B aligned)
#define OFF_APRE 0u
#define OFF_BPRE 1114112u
#define OFF_WL   2228224u
#define OFF_EW2T 2230400u
#define OFF_CW1T 2265216u
#define OFF_MIP  2273408u
#define OFF_COP  2797696u

// Fast SiLU: x * rcp(1 + exp(-x)).  v_exp_f32 + v_rcp_f32 instead of the
// IEEE divide expansion (v_div_scale/v_div_fmas/v_div_fixup, ~8 VALU ops).
// Result feeds bf16 (8-bit mantissa) or loose-tolerance f32 paths, so the
// ~1ulp rcp is plenty.
__device__ __forceinline__ float silu_f(float x) {
    return x * __builtin_amdgcn_rcpf(1.0f + __expf(-x));
}

// ---------------------------------------------------------------------------
// Precompute A_pre[i][c] = feats[i]@eW1[0:128][c] + eb1[c]  (pad c>=514 -> 0)
//            B_pre[i][c] = feats[i]@eW1[128:256][c]         (pad -> 0)
// ---------------------------------------------------------------------------
__launch_bounds__(256)
__global__ void egnn_pre(const float* __restrict__ feats,
                         const float* __restrict__ eW1,
                         const float* __restrict__ eb1,
                         float* __restrict__ A_pre,
                         float* __restrict__ B_pre) {
    __shared__ float fs[DIM];
    const int i = blockIdx.x;
    const int t = threadIdx.x;
    if (t < DIM) fs[t] = feats[i * DIM + t];
    __syncthreads();
    for (int c = t; c < H1P; c += 256) {
        float a = 0.f, b = 0.f;
        if (c < H1) {
            a = eb1[c];
            for (int k = 0; k < DIM; ++k) {
                a = fmaf(fs[k], eW1[k * H1 + c], a);
                b = fmaf(fs[k], eW1[(DIM + k) * H1 + c], b);
            }
        }
        A_pre[i * H1P + c] = a;
        B_pre[i * H1P + c] = b;
    }
}

// ---------------------------------------------------------------------------
// Convert: w_last (eW1 row 256, zero-padded), eW2 -> bf16 transposed [32][544],
//          cW1 -> bf16 transposed [128][32]
// ---------------------------------------------------------------------------
__launch_bounds__(256)
__global__ void egnn_cvt(const float* __restrict__ eW1,
                         const float* __restrict__ eW2,
                         const float* __restrict__ cW1,
                         float* __restrict__ wl,
                         bf16_t* __restrict__ eW2T,
                         bf16_t* __restrict__ cW1T) {
    const int g = blockIdx.x * 256 + threadIdx.x;
    if (g < H1P) {
        wl[g] = (g < H1) ? eW1[256 * H1 + g] : 0.f;
    } else if (g < H1P + MD * H1P) {
        const int u = g - H1P;
        const int n = u / H1P, k = u % H1P;
        eW2T[n * H1P + k] = (k < H1) ? (bf16_t)eW2[k * MD + n] : (bf16_t)0.f;
    } else if (g < H1P + MD * H1P + 128 * MD) {
        const int u = g - H1P - MD * H1P;
        const int n = u / MD, k = u % MD;
        cW1T[n * MD + k] = (bf16_t)cW1[k * 128 + n];
    }
}

// ---------------------------------------------------------------------------
// Edge kernel: grid (32 i-tiles, 8 j-chunks), 256 threads = 8 waves.
// Wave w handles j in {jc*64 + w*8 + t}.  Per j: 34+8 bf16 WMMAs.
// ---------------------------------------------------------------------------
__launch_bounds__(256)
__global__ void egnn_edge(const float* __restrict__ coors,
                          const float* __restrict__ eb2,
                          const float* __restrict__ cb1,
                          const float* __restrict__ cW2,
                          const float* __restrict__ cb2,
                          const float* __restrict__ A_pre,
                          const float* __restrict__ B_pre,
                          const float* __restrict__ wl,
                          const bf16_t* __restrict__ eW2T,
                          const bf16_t* __restrict__ cW1T,
                          float* __restrict__ mi_part,
                          float* __restrict__ coor_part) {
    __shared__ float Alds[16 * H1P];     // 34816 B
    __shared__ float wlS[H1P];           //  2176 B
    __shared__ __align__(32) char scr[18432]; // per-wave bf16 scratch, later f32 partials

    const int tid  = threadIdx.x;
    const int lane = tid & 31;
    const int wave = tid >> 5;
    const int row  = lane & 15;   // A-fragment row (M index)
    const int half = lane >> 4;   // lane group
    const int col  = lane & 15;   // C/D column within 16-block
    const int i0 = blockIdx.x * 16;
    const int jc = blockIdx.y;

    for (int t = tid; t < 16 * H1P; t += 256)
        Alds[t] = A_pre[(i0 + (t / H1P)) * H1P + (t % H1P)];
    for (int t = tid; t < H1P; t += 256)
        wlS[t] = wl[t];
    __syncthreads();

    const float cix = coors[(i0 + row) * 3 + 0];
    const float ciy = coors[(i0 + row) * 3 + 1];
    const float ciz = coors[(i0 + row) * 3 + 2];
    const float e2a = eb2[col];
    const float e2b = eb2[col + 16];
    float cb1v[8], cw2v[8];
#pragma unroll
    for (int nb = 0; nb < 8; ++nb) {
        const int n2 = col + 16 * nb;
        cb1v[nb] = cb1[n2];
        cw2v[nb] = cW2[n2];
    }
    const float cb2s = cb2[0];

    float macc0[8], macc1[8], caccx[8], caccy[8], caccz[8];
#pragma unroll
    for (int r = 0; r < 8; ++r) {
        macc0[r] = 0.f; macc1[r] = 0.f;
        caccx[r] = 0.f; caccy[r] = 0.f; caccz[r] = 0.f;
    }

    bf16_t* msc = (bf16_t*)(scr + wave * 1024);  // 16x32 bf16

    for (int t = 0; t < 8; ++t) {
        const int j = jc * 64 + wave * 8 + t;
        const float cjx = coors[j * 3 + 0];
        const float cjy = coors[j * 3 + 1];
        const float cjz = coors[j * 3 + 2];
        const float dx = cix - cjx, dy = ciy - cjy, dz = ciz - cjz;
        const float rd = dx * dx + dy * dy + dz * dz;
        const float* __restrict__ Bj = B_pre + (size_t)j * H1P;
        const float* __restrict__ Ar = Alds + row * H1P;

        v8f c0, c1;
#pragma unroll
        for (int r = 0; r < 8; ++r) { c0[r] = e2a; c1[r] = e2b; }

        // m_ij = silu(h1) @ eW2 :  M=16 edges, K=544(pad), N=32
        for (int ks = 0; ks < KSTEPS; ++ks) {
            const int k0 = ks * 32 + half * 16;
            v16bf a;
#pragma unroll
            for (int e = 0; e < 8; ++e) {
                const int ka = k0 + e;
                const int kb = k0 + 8 + e;
                a[e]     = (bf16_t)silu_f(Ar[ka] + Bj[ka] + rd * wlS[ka]);
                a[e + 8] = (bf16_t)silu_f(Ar[kb] + Bj[kb] + rd * wlS[kb]);
            }
            const int kB = ks * 32 + half * 16;
            v16bf b0 = *(const v16bf*)(eW2T + (size_t)col * H1P + kB);
            v16bf b1 = *(const v16bf*)(eW2T + (size_t)(col + 16) * H1P + kB);
            c0 = __builtin_amdgcn_wmma_f32_16x16x32_bf16(false, a, false, b0,
                                                         (short)0, c0, false, false);
            c1 = __builtin_amdgcn_wmma_f32_16x16x32_bf16(false, a, false, b1,
                                                         (short)0, c1, false, false);
        }

        // SiLU -> m_ij ; accumulate m_i ; stage bf16 m_ij for next GEMM
#pragma unroll
        for (int r = 0; r < 8; ++r) {
            const float m0 = silu_f(c0[r]);
            const float m1 = silu_f(c1[r]);
            macc0[r] += m0;
            macc1[r] += m1;
            const int m = r + half * 8;
            msc[m * 32 + col]      = (bf16_t)m0;
            msc[m * 32 + col + 16] = (bf16_t)m1;
        }

        // re-gather as A fragment (16x32)
        v16bf a2;
#pragma unroll
        for (int e = 0; e < 8; ++e) {
            a2[e]     = msc[row * 32 + half * 16 + e];
            a2[e + 8] = msc[row * 32 + half * 16 + 8 + e];
        }

        // h2 = silu(m @ cW1 + cb1) ; w = h2 @ cW2 + cb2
        float pr[8];
#pragma unroll
        for (int r = 0; r < 8; ++r) pr[r] = 0.f;
#pragma unroll
        for (int nb = 0; nb < 8; ++nb) {
            v8f cc;
#pragma unroll
            for (int r = 0; r < 8; ++r) cc[r] = cb1v[nb];
            v16bf b2 = *(const v16bf*)(cW1T + (size_t)(col + 16 * nb) * MD + half * 16);
            v8f h = __builtin_amdgcn_wmma_f32_16x16x32_bf16(false, a2, false, b2,
                                                            (short)0, cc, false, false);
#pragma unroll
            for (int r = 0; r < 8; ++r) pr[r] += silu_f(h[r]) * cw2v[nb];
        }
#pragma unroll
        for (int r = 0; r < 8; ++r) {
            pr[r] += __shfl_xor(pr[r], 1, 32);
            pr[r] += __shfl_xor(pr[r], 2, 32);
            pr[r] += __shfl_xor(pr[r], 4, 32);
            pr[r] += __shfl_xor(pr[r], 8, 32);
        }
        if (col == 0) {
#pragma unroll
            for (int r = 0; r < 8; ++r) {
                const int i = i0 + r + half * 8;
                const float w = pr[r] + cb2s;
                const float ddx = coors[i * 3 + 0] - cjx;
                const float ddy = coors[i * 3 + 1] - cjy;
                const float ddz = coors[i * 3 + 2] - cjz;
                const float nrm = sqrtf(ddx * ddx + ddy * ddy + ddz * ddz);
                const float s = w * __builtin_amdgcn_rcpf(fmaxf(nrm, 1e-8f));
                caccx[r] += s * ddx;
                caccy[r] += s * ddy;
                caccz[r] += s * ddz;
            }
        }
    }

    // ---- deterministic block reduction of per-wave partials ----
    __syncthreads();                       // mscr no longer needed
    float* miL = (float*)scr;              // [8][16*32]  (16384 B)
    float* coL = (float*)scr + 8 * 512;    // [8][16*4]   ( 2048 B)
    for (int idx = tid; idx < 512; idx += 256) coL[idx] = 0.f;
    __syncthreads();
#pragma unroll
    for (int r = 0; r < 8; ++r) {
        const int m = r + half * 8;
        miL[wave * 512 + m * 32 + col]      = macc0[r];
        miL[wave * 512 + m * 32 + col + 16] = macc1[r];
    }
    if (col == 0) {
#pragma unroll
        for (int r = 0; r < 8; ++r) {
            const int m = r + half * 8;
            coL[wave * 64 + m * 4 + 0] = caccx[r];
            coL[wave * 64 + m * 4 + 1] = caccy[r];
            coL[wave * 64 + m * 4 + 2] = caccz[r];
        }
    }
    __syncthreads();
    for (int idx = tid; idx < 512; idx += 256) {
        float s = 0.f;
#pragma unroll
        for (int w = 0; w < 8; ++w) s += miL[w * 512 + idx];
        mi_part[jc * (N_NODES * MD) + (i0 + idx / 32) * MD + (idx & 31)] = s;
    }
    if (tid < 64) {
        float s = 0.f;
#pragma unroll
        for (int w = 0; w < 8; ++w) s += coL[w * 64 + tid];
        coor_part[jc * (N_NODES * 4) + (i0 + (tid >> 2)) * 4 + (tid & 3)] = s;
    }
}

// ---------------------------------------------------------------------------
// Node MLP + outputs (tiny: ~75 MFLOP)
// ---------------------------------------------------------------------------
__launch_bounds__(256)
__global__ void egnn_node(const float* __restrict__ feats,
                          const float* __restrict__ coors,
                          const float* __restrict__ nW1,
                          const float* __restrict__ nb1,
                          const float* __restrict__ nW2,
                          const float* __restrict__ nb2,
                          const float* __restrict__ ln_b,
                          const float* __restrict__ mi_part,
                          const float* __restrict__ coor_part,
                          float* __restrict__ node_out,
                          float* __restrict__ coors_out) {
    __shared__ float nin[DIM + MD];
    __shared__ float h[256];
    const int i = blockIdx.x;
    const int t = threadIdx.x;
    if (t < DIM) {
        nin[t] = feats[i * DIM + t];
    } else if (t < DIM + MD) {
        float s = 0.f;
#pragma unroll
        for (int p = 0; p < 8; ++p) s += mi_part[p * (N_NODES * MD) + i * MD + (t - DIM)];
        nin[t] = s;
    }
    __syncthreads();
    float a = nb1[t];
    for (int k = 0; k < DIM + MD; ++k) a = fmaf(nin[k], nW1[k * 256 + t], a);
    // final-output silu kept in full precision via exact divide
    h[t] = a / (1.0f + __expf(-a));
    __syncthreads();
    if (t < DIM) {
        float o = nb2[t];
        for (int k = 0; k < 256; ++k) o = fmaf(h[k], nW2[k * DIM + t], o);
        node_out[i * DIM + t] = o + feats[i * DIM + t];
    }
    if (t < 3) {
        float s = 0.f;
#pragma unroll
        for (int p = 0; p < 8; ++p) s += coor_part[p * (N_NODES * 4) + i * 4 + t];
        coors_out[i * 3 + t] = ln_b[0] * s + coors[i * 3 + t];
    }
}

extern "C" void kernel_launch(void* const* d_in, const int* in_sizes, int n_in,
                              void* d_out, int out_size, void* d_ws, size_t ws_size,
                              hipStream_t stream) {
    (void)in_sizes; (void)n_in; (void)out_size; (void)ws_size;
    const float* feats = (const float*)d_in[0];
    const float* coors = (const float*)d_in[1];
    const float* eW1   = (const float*)d_in[2];
    const float* eb1   = (const float*)d_in[3];
    const float* eW2   = (const float*)d_in[4];
    const float* eb2   = (const float*)d_in[5];
    const float* cW1   = (const float*)d_in[6];
    const float* cb1   = (const float*)d_in[7];
    const float* cW2   = (const float*)d_in[8];
    const float* cb2   = (const float*)d_in[9];
    const float* nW1   = (const float*)d_in[10];
    const float* nb1   = (const float*)d_in[11];
    const float* nW2   = (const float*)d_in[12];
    const float* nb2   = (const float*)d_in[13];
    const float* ln_b  = (const float*)d_in[15];

    char* ws = (char*)d_ws;
    float*  A_pre     = (float*)(ws + OFF_APRE);
    float*  B_pre     = (float*)(ws + OFF_BPRE);
    float*  wl        = (float*)(ws + OFF_WL);
    bf16_t* eW2T      = (bf16_t*)(ws + OFF_EW2T);
    bf16_t* cW1T      = (bf16_t*)(ws + OFF_CW1T);
    float*  mi_part   = (float*)(ws + OFF_MIP);
    float*  coor_part = (float*)(ws + OFF_COP);

    float* node_out  = (float*)d_out;
    float* coors_out = node_out + N_NODES * DIM;

    egnn_cvt<<<88, 256, 0, stream>>>(eW1, eW2, cW1, wl, eW2T, cW1T);
    egnn_pre<<<N_NODES, 256, 0, stream>>>(feats, eW1, eb1, A_pre, B_pre);
    egnn_edge<<<dim3(32, 8), 256, 0, stream>>>(coors, eb2, cb1, cW2, cb2,
                                               A_pre, B_pre, wl, eW2T, cW1T,
                                               mi_part, coor_part);
    egnn_node<<<N_NODES, 256, 0, stream>>>(feats, coors, nW1, nb1, nW2, nb2,
                                           ln_b, mi_part, coor_part,
                                           node_out, coors_out);
}
